// CrossGSA_66279935311933
// MI455X (gfx1250) — matrix operand order repaired and checked by reading, not compiled, not run on hardware
//
#include <hip/hip_runtime.h>

// ---------------- problem constants ----------------
#define BATCH   2
#define HH      48
#define WW      48
#define NTOK    2304            // 48*48
#define CCH     256
#define NHEADS  8
#define HDIM    32
#define ROWS    4608            // BATCH*NTOK
#define SCALING 0.17677669529663687f   // 32^-0.5
#define LNEPS   1e-6f

typedef __attribute__((ext_vector_type(16))) _Float16 v16h;
typedef __attribute__((ext_vector_type(8)))  float    v8f;
typedef __attribute__((ext_vector_type(4)))  unsigned int u32x4;
typedef __attribute__((ext_vector_type(8)))  int      i32x8;
typedef __attribute__((ext_vector_type(4)))  int      i32x4;

#if defined(__has_builtin)
#  if __has_builtin(__builtin_amdgcn_tensor_load_to_lds) && __has_builtin(__builtin_amdgcn_s_wait_tensorcnt)
#    define HAVE_TDM 1
#  endif
#endif
#ifndef HAVE_TDM
#  define HAVE_TDM 0
#endif

// ---------------- WMMA fragment helpers (wave32, gfx1250 layouts) ----------------
// A (16x32 f16): lane holds row m = lane&15; element e -> K = (e&7) + 8*half + ((e>>3)<<4)
// Per lane: two contiguous 16B runs -> 2x global_load_b128.
__device__ __forceinline__ v16h load_a_f16(const _Float16* __restrict__ A, int ld) {
  int lane = threadIdx.x & 31;
  int hf   = lane >> 4;
  int m    = lane & 15;
  const _Float16* row = A + (size_t)m * ld + 8 * hf;
  v16h a;
#pragma unroll
  for (int e = 0; e < 16; ++e) {
    int k = (e & 7) + ((e >> 3) << 4);
    a[e] = row[k];
  }
  return a;
}

// A fragment from f32 memory (used only for attn-out+lepe in proj GEMM)
__device__ __forceinline__ v16h load_a_f32(const float* __restrict__ A, int ld) {
  int lane = threadIdx.x & 31;
  int hf   = lane >> 4;
  int m    = lane & 15;
  const float* row = A + (size_t)m * ld + 8 * hf;
  v16h a;
#pragma unroll
  for (int e = 0; e < 16; ++e) {
    int k = (e & 7) + ((e >> 3) << 4);
    a[e] = (_Float16)row[k];
  }
  return a;
}

// B (32x16 f16) where memory holds the transpose: Bm[n][k], row stride ld.
// lane owns col n = lane&15; element e -> K = (e&7) + 8*half + ((e>>3)<<4)
__device__ __forceinline__ v16h load_bT_f16(const _Float16* __restrict__ Bm, int ld) {
  int lane = threadIdx.x & 31;
  int hf   = lane >> 4;
  int n    = lane & 15;
  const _Float16* p = Bm + (size_t)n * ld + 8 * hf;
  v16h b;
#pragma unroll
  for (int e = 0; e < 16; ++e) b[e] = p[(e & 7) + ((e >> 3) << 4)];
  return b;
}

__device__ __forceinline__ v8f wmma16(v16h a, v16h b, v8f c) {
  return __builtin_amdgcn_wmma_f32_16x16x32_f16(false, a, false, b, (short)0, c,
                                                false, false);
}

// ---------------- P0a: convert+transpose weights  W[k][n] f32 -> WT[n][k] f16 ----------------
__global__ void prep_w_kernel(const float* __restrict__ w0, const float* __restrict__ w1,
                              const float* __restrict__ w2, const float* __restrict__ w3,
                              _Float16* __restrict__ o0, _Float16* __restrict__ o1,
                              _Float16* __restrict__ o2, _Float16* __restrict__ o3) {
  int idx = blockIdx.x * blockDim.x + threadIdx.x;     // 0..65535
  int which = blockIdx.y;
  const float* W  = which == 0 ? w0 : which == 1 ? w1 : which == 2 ? w2 : w3;
  _Float16*    WT = which == 0 ? o0 : which == 1 ? o1 : which == 2 ? o2 : o3;
  int n = idx & (CCH - 1);
  int k = idx >> 8;
  WT[(size_t)n * CCH + k] = (_Float16)W[(size_t)k * CCH + n];
}

// ---------------- P0b: x f32 -> f16 ----------------
__global__ void prep_x_kernel(const float* __restrict__ x, _Float16* __restrict__ xh) {
  int idx = blockIdx.x * blockDim.x + threadIdx.x;
  if (idx < ROWS * CCH) xh[idx] = (_Float16)x[idx];
}

// ---------------- K1: fused QKV projection (WMMA f16) ----------------
// writes q,k,v f32 [b,head,n,d]; for V also writes f16 transposed copy [b,head,d,n]
__global__ void qkv_gemm_kernel(const _Float16* __restrict__ xh,
                                const _Float16* __restrict__ wqh, const float* __restrict__ bq,
                                const _Float16* __restrict__ wkh, const float* __restrict__ bk,
                                const _Float16* __restrict__ wvh, const float* __restrict__ bv,
                                float* __restrict__ qb, float* __restrict__ kb,
                                float* __restrict__ vb, _Float16* __restrict__ vhT) {
  int wave = blockIdx.x * (blockDim.x >> 5) + (threadIdx.x >> 5);
  int tn    = wave & 15;
  int tmp   = wave >> 4;
  int tm    = tmp % (ROWS / 16);
  int which = tmp / (ROWS / 16);        // 0=q 1=k 2=v
  if (which >= 3) return;
  const _Float16* W    = which == 0 ? wqh : which == 1 ? wkh : wvh;
  const float*    bias = which == 0 ? bq  : which == 1 ? bk  : bv;
  float*          out  = which == 0 ? qb  : which == 1 ? kb  : vb;
  float scale = which == 1 ? SCALING : 1.0f;

  v8f acc = {};
#pragma unroll
  for (int k0 = 0; k0 < CCH; k0 += 32) {
    v16h a = load_a_f16(xh + (size_t)tm * 16 * CCH + k0, CCH);
    v16h b = load_bT_f16(W + (size_t)tn * 16 * CCH + k0, CCH);
    acc = wmma16(a, b, acc);
  }
  int lane = threadIdx.x & 31;
  int hf   = lane >> 4;
  int c    = tn * 16 + (lane & 15);
  int head = c >> 5, d = c & 31;
  float bval = bias[c];
#pragma unroll
  for (int r = 0; r < 8; ++r) {
    int rr = tm * 16 + r + 8 * hf;
    int b_ = rr / NTOK;
    int n  = rr - b_ * NTOK;
    float val = (acc[r] + bval) * scale;
    out[(size_t)((b_ * NHEADS + head) * NTOK + n) * HDIM + d] = val;
    if (which == 2)
      vhT[((size_t)(b_ * NHEADS + head) * HDIM + d) * NTOK + n] = (_Float16)val;
  }
}

// ---------------- K2: rotary on q,k -> f16 copies ----------------
__global__ void rotary_kernel(const float* __restrict__ q, const float* __restrict__ k,
                              const float* __restrict__ sn, const float* __restrict__ cs,
                              _Float16* __restrict__ qh, _Float16* __restrict__ kh) {
  int idx = blockIdx.x * blockDim.x + threadIdx.x;   // pair index
  const int PAIRS = BATCH * NHEADS * NTOK * (HDIM / 2);
  if (idx >= PAIRS) return;
  int d2 = idx & 15;
  int n  = (idx >> 4) % NTOK;
  int off = idx * 2;
  float s0 = sn[n * HDIM + 2 * d2], s1 = sn[n * HDIM + 2 * d2 + 1];
  float c0 = cs[n * HDIM + 2 * d2], c1 = cs[n * HDIM + 2 * d2 + 1];
  float2 qv = *(const float2*)(q + off);
  float2 kv = *(const float2*)(k + off);
  qh[off]     = (_Float16)(qv.x * c0 - qv.y * s0);
  qh[off + 1] = (_Float16)(qv.y * c1 + qv.x * s1);
  kh[off]     = (_Float16)(kv.x * c0 - kv.y * s0);
  kh[off + 1] = (_Float16)(kv.y * c1 + kv.x * s1);
}

// ---------------- K3: 5x5 depthwise conv (LEPE) -> lp [b][n][c] ----------------
__global__ void lepe_kernel(const float* __restrict__ v, const float* __restrict__ wgt,
                            const float* __restrict__ bias, float* __restrict__ out) {
  int idx = blockIdx.x * blockDim.x + threadIdx.x;
  if (idx >= ROWS * CCH) return;
  int c  = idx & (CCH - 1);
  int bn = idx >> 8;
  int b  = bn / NTOK;
  int n  = bn - b * NTOK;
  int h = n / WW, w = n - h * WW;
  int head = c >> 5, d = c & 31;
  const float* vb = v + (size_t)((b * NHEADS + head) * NTOK) * HDIM + d;
  const float* wc = wgt + c * 25;
  float acc = bias[c];
#pragma unroll
  for (int dy = 0; dy < 5; ++dy) {
    int hy = h + dy - 2;
    if ((unsigned)hy >= (unsigned)HH) continue;
#pragma unroll
    for (int dx = 0; dx < 5; ++dx) {
      int wx = w + dx - 2;
      if ((unsigned)wx >= (unsigned)WW) continue;
      acc += wc[dy * 5 + dx] * vb[(size_t)(hy * WW + wx) * HDIM];
    }
  }
  out[idx] = acc;
}

#if HAVE_TDM
// TDM: DMA a 16-row x 32-col f32 tile (row stride NTOK floats) into LDS.
// Descriptor per CDNA5 ISA (2D tensor -> groups 2/3 disabled).
// This toolchain's builtin takes 6 args: (u32x4, i32x8, i32x4, i32x4, i32x8, cpol).
__device__ __forceinline__ void tdm_load_mask_tile(const float* gsrc, float* ldst) {
  unsigned long long ga = (unsigned long long)(size_t)gsrc;
  unsigned lo = (unsigned)(size_t)ldst;          // low 32 bits of generic ptr = LDS offset
  u32x4 g0;
  g0[0] = 1u;                                    // count=1 valid user descriptor
  g0[1] = lo;                                    // lds_addr
  g0[2] = (unsigned)ga;                          // global_addr[31:0]
  g0[3] = (unsigned)(ga >> 32) | 0x80000000u;    // global_addr[56:32] | type=2<<30
  i32x8 g1;
  g1[0] = (int)(2u << 16);                       // data_size=4B; no multicast/pad/iterate
  g1[1] = (int)((NTOK & 0xFFFF) << 16);          // tensor_dim0[15:0]
  g1[2] = (int)((NTOK >> 16) | ((NTOK & 0xFFFF) << 16));   // dim0 hi | tensor_dim1 lo
  g1[3] = (int)((NTOK >> 16) | (32 << 16));      // dim1 hi | tile_dim0=32
  g1[4] = 16;                                    // tile_dim1=16, tile_dim2=0
  g1[5] = NTOK;                                  // tensor_dim0_stride lo32
  g1[6] = 0;                                     // stride0 hi | stride1 lo
  g1[7] = 0;
  i32x4 g2 = {0, 0, 0, 0};
  i32x4 g3 = {0, 0, 0, 0};
  i32x8 g4 = {0, 0, 0, 0, 0, 0, 0, 0};
  __builtin_amdgcn_tensor_load_to_lds(g0, g1, g2, g3, g4, 0);
}
#endif

// ---------------- K4: masked attention, flash-style online softmax ----------------
// One wave per (b, head, 16-query tile); 32 keys/iteration; mask tile double-buffered
// through LDS by the Tensor Data Mover; accumulates into lp (which holds LEPE).
__global__ void attn_kernel(const _Float16* __restrict__ qh, const _Float16* __restrict__ kh,
                            const _Float16* __restrict__ vhT, const float* __restrict__ mask,
                            float* __restrict__ out) {
  __shared__ _Float16 psm[4][16 * 32];        // P relayout staging, per wave
#if HAVE_TDM
  __shared__ float mbuf[4][2][16 * 32];       // double-buffered mask tiles, per wave
#endif
  int wid  = threadIdx.x >> 5;
  int wave = blockIdx.x * 4 + wid;
  int b    = wave & 1;            // batch fastest: both batches of (head,qt) share mask L2 lines
  int t    = wave >> 1;
  int head = t & 7;
  int qt   = t >> 3;              // 0..143
  int lane = threadIdx.x & 31;
  int hf   = lane >> 4;
  int ln   = lane & 15;
  int bh   = b * NHEADS + head;

  const _Float16* qp = qh  + (size_t)(bh * NTOK + qt * 16) * HDIM;
  const _Float16* kp = kh  + (size_t)(bh * NTOK) * HDIM;
  const _Float16* vp = vhT + (size_t)bh * HDIM * NTOK;
  const float*    mp = mask + (size_t)head * NTOK * NTOK + (size_t)(qt * 16) * NTOK;

  v16h aq = load_a_f16(qp, HDIM);
  float mrow[8], lrow[8];
#pragma unroll
  for (int r = 0; r < 8; ++r) { mrow[r] = -1e30f; lrow[r] = 0.f; }
  v8f o0 = {}, o1 = {};
  v8f zero = {};
  _Float16* lds = &psm[wid][0];

#if HAVE_TDM
  float* mb0 = &mbuf[wid][0][0];
  float* mb1 = &mbuf[wid][1][0];
  tdm_load_mask_tile(mp, mb0);                  // prefetch first tile
#endif
  int cur = 0;

  for (int kbase = 0; kbase < NTOK; kbase += 32) {
    // scores: S = Q @ K^T ; per-lane contiguous f16 fragment loads
    v8f s0 = wmma16(aq, load_bT_f16(kp + (size_t)kbase * HDIM, HDIM), zero);
    v8f s1 = wmma16(aq, load_bT_f16(kp + (size_t)(kbase + 16) * HDIM, HDIM), zero);

#if HAVE_TDM
    // drain our LDS reads of the buffer we are about to overwrite, then prefetch next
    asm volatile("s_wait_dscnt 0" ::: "memory");
    if (kbase + 32 < NTOK) {
      tdm_load_mask_tile(mp + kbase + 32, cur ? mb0 : mb1);
      __builtin_amdgcn_s_wait_tensorcnt(1);     // current tile has landed
    } else {
      __builtin_amdgcn_s_wait_tensorcnt(0);
    }
    const float* mtile = cur ? mb1 : mb0;
#endif

#pragma unroll
    for (int r = 0; r < 8; ++r) {
      int m = r + 8 * hf;                       // query row this lane owns in C/D
#if HAVE_TDM
      float a0 = mtile[m * 32 + ln];
      float a1 = mtile[m * 32 + 16 + ln];
#else
      const float* mr = mp + (size_t)m * NTOK + kbase;
      float a0 = mr[ln];
      float a1 = mr[16 + ln];
#endif
      float s0r = s0[r] + a0;
      float s1r = s1[r] + a1;
      float tmax = fmaxf(s0r, s1r);
      tmax = fmaxf(tmax, __shfl_xor(tmax, 1, 16));
      tmax = fmaxf(tmax, __shfl_xor(tmax, 2, 16));
      tmax = fmaxf(tmax, __shfl_xor(tmax, 4, 16));
      tmax = fmaxf(tmax, __shfl_xor(tmax, 8, 16));
      float newm  = fmaxf(mrow[r], tmax);
      float alpha = __expf(mrow[r] - newm);
      float p0 = __expf(s0r - newm);
      float p1 = __expf(s1r - newm);
      float rs = p0 + p1;
      rs += __shfl_xor(rs, 1, 16);
      rs += __shfl_xor(rs, 2, 16);
      rs += __shfl_xor(rs, 4, 16);
      rs += __shfl_xor(rs, 8, 16);
      lrow[r] = lrow[r] * alpha + rs;
      mrow[r] = newm;
      o0[r] = o0[r] * alpha;
      o1[r] = o1[r] * alpha;
      lds[m * 32 + ln]      = (_Float16)p0;     // stage P for A-fragment relayout
      lds[m * 32 + 16 + ln] = (_Float16)p1;
    }
    // wave-local: DS ops are in-order per wave; wait for stores before relayout loads
    asm volatile("s_wait_dscnt 0" ::: "memory");

    v16h pa;                                    // P as A fragment (16 queries x 32 keys)
    {
      const _Float16* prow = lds + ln * 32 + 8 * hf;
#pragma unroll
      for (int e = 0; e < 16; ++e)
        pa[e] = prow[(e & 7) + ((e >> 3) << 4)];
    }
    // PV: B fragments from transposed f16 V -> per-lane contiguous loads
    o0 = wmma16(pa, load_bT_f16(vp + (size_t)0  * NTOK + kbase, NTOK), o0);
    o1 = wmma16(pa, load_bT_f16(vp + (size_t)16 * NTOK + kbase, NTOK), o1);
    cur ^= 1;
  }

  float* op = out + ((size_t)(b * NTOK + qt * 16)) * CCH + head * HDIM;
#pragma unroll
  for (int r = 0; r < 8; ++r) {
    int m = r + 8 * hf;
    float inv = 1.0f / lrow[r];
    op[(size_t)m * CCH + ln]      += o0[r] * inv;   // += LEPE already there
    op[(size_t)m * CCH + 16 + ln] += o1[r] * inv;
  }
}

// ---------------- K5: output projection + input residual (WMMA) -> y (in d_out) ----------------
__global__ void proj_kernel(const float* __restrict__ a_in, const float* __restrict__ x,
                            const _Float16* __restrict__ woh, const float* __restrict__ bo,
                            float* __restrict__ y) {
  int wave = blockIdx.x * (blockDim.x >> 5) + (threadIdx.x >> 5);
  int tn = wave & 15;
  int tm = wave >> 4;
  if (tm >= ROWS / 16) return;
  int lane = threadIdx.x & 31;
  int hf   = lane >> 4;
  int m    = lane & 15;
  v8f acc = {};
#pragma unroll
  for (int k0 = 0; k0 < CCH; k0 += 32) {
    v16h a = load_a_f32(a_in + (size_t)tm * 16 * CCH + k0, CCH);
    v16h b = load_bT_f16(woh + (size_t)tn * 16 * CCH + k0, CCH);
    acc = wmma16(a, b, acc);
  }
  int c = tn * 16 + m;
  float bval = bo[c];
#pragma unroll
  for (int r = 0; r < 8; ++r) {
    int rr = tm * 16 + r + 8 * hf;
    y[(size_t)rr * CCH + c] = x[(size_t)rr * CCH + c] + acc[r] + bval;
  }
}

// ---------------- K6: out = y + LayerNorm(y), one wave per 256-wide row, in place ----------------
__global__ void ln_kernel(float* __restrict__ y, const float* __restrict__ g,
                          const float* __restrict__ bb) {
  int wave = blockIdx.x * (blockDim.x >> 5) + (threadIdx.x >> 5);
  if (wave >= ROWS) return;
  int lane = threadIdx.x & 31;
  float* yr = y + (size_t)wave * CCH;
  float vals[8];
  float s = 0.f, s2 = 0.f;
#pragma unroll
  for (int e = 0; e < 8; ++e) {
    float t = yr[e * 32 + lane];
    vals[e] = t; s += t; s2 += t * t;
  }
#pragma unroll
  for (int msk = 16; msk >= 1; msk >>= 1) {
    s  += __shfl_xor(s,  msk, 32);
    s2 += __shfl_xor(s2, msk, 32);
  }
  float mu  = s * (1.0f / CCH);
  float var = s2 * (1.0f / CCH) - mu * mu;
  float inv = rsqrtf(var + LNEPS);
#pragma unroll
  for (int e = 0; e < 8; ++e) {
    int c = e * 32 + lane;
    yr[c] = vals[e] + (vals[e] - mu) * inv * g[c] + bb[c];
  }
}

// ---------------- launcher ----------------
extern "C" void kernel_launch(void* const* d_in, const int* in_sizes, int n_in,
                              void* d_out, int out_size, void* d_ws, size_t ws_size,
                              hipStream_t stream) {
  (void)in_sizes; (void)n_in; (void)out_size; (void)ws_size;
  const float* x      = (const float*)d_in[0];
  const float* sn     = (const float*)d_in[1];
  const float* cs     = (const float*)d_in[2];
  const float* mask   = (const float*)d_in[3];
  const float* wq     = (const float*)d_in[4];
  const float* bq     = (const float*)d_in[5];
  const float* wk     = (const float*)d_in[6];
  const float* bk     = (const float*)d_in[7];
  const float* wv     = (const float*)d_in[8];
  const float* bv     = (const float*)d_in[9];
  const float* lepe_w = (const float*)d_in[10];
  const float* lepe_b = (const float*)d_in[11];
  const float* wo     = (const float*)d_in[12];
  const float* bo     = (const float*)d_in[13];
  const float* ln_g   = (const float*)d_in[14];
  const float* ln_b   = (const float*)d_in[15];
  float* out = (float*)d_out;

  size_t sz = (size_t)ROWS * CCH;           // 1,179,648 elements per activation buffer
  float*    qb  = (float*)d_ws;             // q f32 [b,head,n,d]
  float*    kb  = qb + sz;                  // k f32 [b,head,n,d] (pre-scaled)
  float*    vb  = kb + sz;                  // v f32 [b,head,n,d]
  float*    lp  = vb + sz;                  // lepe then += attn out  [b,n,c]
  _Float16* xh  = (_Float16*)(lp + sz);     // x f16
  _Float16* qh  = xh + sz;                  // rotated q f16
  _Float16* kh  = qh + sz;                  // rotated k f16
  _Float16* vhT = kh + sz;                  // v f16 transposed [b,head,d,n]
  _Float16* wqh = vhT + sz;                 // weights f16 transposed [n][k]
  _Float16* wkh = wqh + (size_t)CCH * CCH;
  _Float16* wvh = wkh + (size_t)CCH * CCH;
  _Float16* woh = wvh + (size_t)CCH * CCH;

  prep_w_kernel<<<dim3(256, 4), 256, 0, stream>>>(wq, wk, wv, wo, wqh, wkh, wvh, woh);
  prep_x_kernel<<<4608, 256, 0, stream>>>(x, xh);
  qkv_gemm_kernel<<<3456, 128, 0, stream>>>(xh, wqh, bq, wkh, bk, wvh, bv, qb, kb, vb, vhT);
  rotary_kernel<<<2304, 256, 0, stream>>>(qb, kb, sn, cs, qh, kh);
  lepe_kernel<<<4608, 256, 0, stream>>>(vb, lepe_w, lepe_b, lp);
  attn_kernel<<<576, 128, 0, stream>>>(qh, kh, vhT, mask, lp);
  proj_kernel<<<1152, 128, 0, stream>>>(lp, x, woh, bo, out);
  ln_kernel<<<1152, 128, 0, stream>>>(out, ln_g, ln_b);
}
